// AudioMamba_37211596653127
// MI455X (gfx1250) — compile-verified, matrix operand order checked by reading, not run on hardware
//
#include <hip/hip_runtime.h>
#include <hip/hip_bf16.h>
#include <math.h>

#define D_MODEL 768
#define D_INNER 1536
#define D_STATE 16
#define DT_RANK 48
#define SEQ 512
#define LTOT 513
#define TOKPOS 256
#define DEPTH 24
#define NUM_CLASSES 527

typedef __attribute__((ext_vector_type(16))) __bf16 v16bf;
typedef __attribute__((ext_vector_type(8)))  float  v8f;

// ---------------------------------------------------------------------------
// bf16-WMMA GEMM:  C[M][N] = epilogue( A[M][K] * W[N][K]^T )
//   One wave computes a 16(M) x 64(N) strip: the A fragment is reused by 4
//   independent v_wmma_f32_16x16x32_bf16 issues per 32-wide K step.
//   Interior strips (full M-tile, N strip in range, K%32==0) take a
//   wave-uniform fast path with zero masking/branching: float2 loads ->
//   v_cvt_pk_bf16_f32 -> 4 WMMAs.  Edge strips use clamped addresses +
//   value selects (no EXEC divergence).
//   MODEA (template): 0 = A as-is; 1 = A := 0.5*(A[m] + Arev[M-1-m])
//   modeE: 0 = none; 1 = +bias[n]; 2 = softplus(acc + bias[n])
// ---------------------------------------------------------------------------
template <int MODEA>
__global__ void __launch_bounds__(128)
gemm_bf16_wmma(const float* __restrict__ A, const float* __restrict__ Arev,
               const float* __restrict__ W, const float* __restrict__ bias,
               float* __restrict__ C, int M, int N, int K, int lda, int modeE)
{
    const int wave   = threadIdx.x >> 5;
    const int lane   = threadIdx.x & 31;
    const int stripsN = (N + 63) >> 6;
    const int tilesM  = (M + 15) >> 4;
    const int strip   = blockIdx.x * 4 + wave;
    if (strip >= tilesM * stripsN) return;          // wave-uniform exit
    const int tm = strip / stripsN;
    const int sn = strip % stripsN;
    const int g  = lane >> 4;                       // lane group (K-half select)
    const int r  = lane & 15;                       // row/col within tile

    const int  m   = tm * 16 + r;
    const int  mc  = (m < M) ? m : (M - 1);
    const bool mok = (m < M);
    const float* __restrict__ ap  = A + (size_t)mc * lda;
    const float* __restrict__ arp = (MODEA == 1) ? (Arev + (size_t)(M - 1 - mc) * lda) : ap;

    const float* wp[4];
    bool nok[4];
#pragma unroll
    for (int t = 0; t < 4; ++t) {
        const int n = sn * 64 + t * 16 + r;
        nok[t] = (n < N);
        wp[t]  = W + (size_t)(nok[t] ? n : (N - 1)) * K;
    }

    v8f acc0 = {}, acc1 = {}, acc2 = {}, acc3 = {};
    const int kblocks = (K + 31) >> 5;

    // Wave-uniform: N is always a multiple of 16 here, so (n < N) is
    // lane-invariant; full fast path when no masking is ever needed.
    const bool interior = ((tm * 16 + 16) <= M) && ((sn * 64 + 64) <= N) &&
                          ((K & 31) == 0);

    if (interior) {
        for (int b = 0; b < kblocks; ++b) {
            const int k0 = b << 5;
            __builtin_prefetch(ap + k0 + 32, 0, 1); // global_prefetch_b8
            v16bf af, bf0, bf1, bf2, bf3;
#pragma unroll
            for (int j = 0; j < 8; ++j) {
                // ISA 7.12.2: 16-bit A 16x32 — VGPR j / group g holds K pair p
                const int p = (j < 4) ? (j + 4 * g) : (j + 4 + 4 * g);
                const int k = k0 + 2 * p;
                float2 av = *(const float2*)(ap + k);
                if (MODEA == 1) {
                    const float2 rv = *(const float2*)(arp + k);
                    av.x = 0.5f * (av.x + rv.x);
                    av.y = 0.5f * (av.y + rv.y);
                }
                af[2 * j]     = (__bf16)av.x;
                af[2 * j + 1] = (__bf16)av.y;
                const float2 b0 = *(const float2*)(wp[0] + k);
                const float2 b1 = *(const float2*)(wp[1] + k);
                const float2 b2 = *(const float2*)(wp[2] + k);
                const float2 b3 = *(const float2*)(wp[3] + k);
                bf0[2 * j] = (__bf16)b0.x; bf0[2 * j + 1] = (__bf16)b0.y;
                bf1[2 * j] = (__bf16)b1.x; bf1[2 * j + 1] = (__bf16)b1.y;
                bf2[2 * j] = (__bf16)b2.x; bf2[2 * j + 1] = (__bf16)b2.y;
                bf3[2 * j] = (__bf16)b3.x; bf3[2 * j + 1] = (__bf16)b3.y;
            }
            acc0 = __builtin_amdgcn_wmma_f32_16x16x32_bf16(false, af, false, bf0, (short)0, acc0, false, false);
            acc1 = __builtin_amdgcn_wmma_f32_16x16x32_bf16(false, af, false, bf1, (short)0, acc1, false, false);
            acc2 = __builtin_amdgcn_wmma_f32_16x16x32_bf16(false, af, false, bf2, (short)0, acc2, false, false);
            acc3 = __builtin_amdgcn_wmma_f32_16x16x32_bf16(false, af, false, bf3, (short)0, acc3, false, false);
        }
    } else {
        for (int b = 0; b < kblocks; ++b) {
            const int k0 = b << 5;
            v16bf af, bf0, bf1, bf2, bf3;
#pragma unroll
            for (int j = 0; j < 8; ++j) {
                const int  p   = (j < 4) ? (j + 4 * g) : (j + 4 + 4 * g);
                const int  k   = k0 + 2 * p;
                const bool kin = (k < K);           // K even: pairs all-or-none
                const int  kc  = kin ? k : 0;

                float2 av = *(const float2*)(ap + kc);
                if (MODEA == 1) {
                    const float2 rv = *(const float2*)(arp + kc);
                    av.x = 0.5f * (av.x + rv.x);
                    av.y = 0.5f * (av.y + rv.y);
                }
                const bool va = mok & kin;
                af[2 * j]     = (__bf16)(va ? av.x : 0.f);
                af[2 * j + 1] = (__bf16)(va ? av.y : 0.f);

                const float2 b0 = *(const float2*)(wp[0] + kc);
                const float2 b1 = *(const float2*)(wp[1] + kc);
                const float2 b2 = *(const float2*)(wp[2] + kc);
                const float2 b3 = *(const float2*)(wp[3] + kc);
                const bool v0 = nok[0] & kin, v1 = nok[1] & kin;
                const bool v2 = nok[2] & kin, v3 = nok[3] & kin;
                bf0[2 * j] = (__bf16)(v0 ? b0.x : 0.f); bf0[2 * j + 1] = (__bf16)(v0 ? b0.y : 0.f);
                bf1[2 * j] = (__bf16)(v1 ? b1.x : 0.f); bf1[2 * j + 1] = (__bf16)(v1 ? b1.y : 0.f);
                bf2[2 * j] = (__bf16)(v2 ? b2.x : 0.f); bf2[2 * j + 1] = (__bf16)(v2 ? b2.y : 0.f);
                bf3[2 * j] = (__bf16)(v3 ? b3.x : 0.f); bf3[2 * j + 1] = (__bf16)(v3 ? b3.y : 0.f);
            }
            acc0 = __builtin_amdgcn_wmma_f32_16x16x32_bf16(false, af, false, bf0, (short)0, acc0, false, false);
            acc1 = __builtin_amdgcn_wmma_f32_16x16x32_bf16(false, af, false, bf1, (short)0, acc1, false, false);
            acc2 = __builtin_amdgcn_wmma_f32_16x16x32_bf16(false, af, false, bf2, (short)0, acc2, false, false);
            acc3 = __builtin_amdgcn_wmma_f32_16x16x32_bf16(false, af, false, bf3, (short)0, acc3, false, false);
        }
    }

#pragma unroll
    for (int t = 0; t < 4; ++t) {
        const v8f acc = (t == 0) ? acc0 : (t == 1) ? acc1 : (t == 2) ? acc2 : acc3;
        const int nn = sn * 64 + t * 16 + r;
#pragma unroll
        for (int v = 0; v < 8; ++v) {
            const int mm = tm * 16 + v + 8 * g;     // C/D layout: VGPR v -> M
            if (mm < M && nn < N) {
                float val = acc[v];
                if (modeE == 1) {
                    val += bias[nn];
                } else if (modeE == 2) {
                    val += bias[nn];
                    val = (val > 20.f) ? val : log1pf(__expf(val)); // softplus
                }
                C[(size_t)mm * N + nn] = val;
            }
        }
    }
}

// ---------------------------------------------------------------------------
// im2col for the 16x16/stride-16 patch embed: A[t=gy*64+gx][k=i*16+j]
// ---------------------------------------------------------------------------
__global__ void __launch_bounds__(256)
im2col_patch(const float* __restrict__ x, float* __restrict__ Ap)
{
    const int idx = blockIdx.x * 256 + threadIdx.x;
    if (idx >= SEQ * 256) return;
    const int t = idx >> 8, k = idx & 255;
    const int gy = t >> 6, gx = t & 63;
    const int i = k >> 4, j = k & 15;
    Ap[idx] = x[(size_t)(gy * 16 + i) * 1024 + gx * 16 + j];
}

// ---------------------------------------------------------------------------
// Insert CLS token at TOKPOS, add pos_embed, zero the residual stream.
// ---------------------------------------------------------------------------
__global__ void __launch_bounds__(256)
assemble_tokens(const float* __restrict__ tokens, const float* __restrict__ cls,
                const float* __restrict__ pos, float* __restrict__ hidden,
                float* __restrict__ residual)
{
    const int idx = blockIdx.x * 256 + threadIdx.x;
    if (idx >= LTOT * D_MODEL) return;
    const int l = idx / D_MODEL, c = idx % D_MODEL;
    float v;
    if (l < TOKPOS)       v = tokens[(size_t)l * D_MODEL + c];
    else if (l == TOKPOS) v = cls[c];
    else                  v = tokens[(size_t)(l - 1) * D_MODEL + c];
    hidden[idx]   = v + pos[idx];
    residual[idx] = 0.f;
}

// ---------------------------------------------------------------------------
// residual += hidden;  hout = rmsnorm(residual) * nw     (one block per token)
// ---------------------------------------------------------------------------
__global__ void __launch_bounds__(256)
resid_rmsnorm(const float* __restrict__ hidden, float* __restrict__ residual,
              const float* __restrict__ nw, float* __restrict__ hout)
{
    __shared__ float red[256];
    const int l = blockIdx.x;
    float ss = 0.f;
    for (int c = threadIdx.x; c < D_MODEL; c += 256) {
        const float r = residual[(size_t)l * D_MODEL + c] + hidden[(size_t)l * D_MODEL + c];
        residual[(size_t)l * D_MODEL + c] = r;
        ss += r * r;
    }
    red[threadIdx.x] = ss;
    __syncthreads();
    for (int s = 128; s > 0; s >>= 1) {
        if (threadIdx.x < s) red[threadIdx.x] += red[threadIdx.x + s];
        __syncthreads();
    }
    const float scale = rsqrtf(red[0] / (float)D_MODEL + 1e-5f);
    for (int c = threadIdx.x; c < D_MODEL; c += 256)
        hout[(size_t)l * D_MODEL + c] = residual[(size_t)l * D_MODEL + c] * scale * nw[c];
}

// ---------------------------------------------------------------------------
// Depthwise causal conv (D_CONV=4) + SiLU for both directions at once.
// xz is [L][3072] token-major (x = cols 0..1535).  Backward output is stored
// in reversed sequence order (index j = L-1-l), so the scan runs "forward".
// ---------------------------------------------------------------------------
__global__ void __launch_bounds__(256)
conv_silu(const float* __restrict__ xz, const float* __restrict__ cw,
          const float* __restrict__ cb, float* __restrict__ xcf,
          float* __restrict__ xcb, int L)
{
    const int idx = blockIdx.x * 256 + threadIdx.x;
    if (idx >= L * D_INNER) return;
    const int d = idx % D_INNER, l = idx / D_INNER;

    const float* wf = cw + (size_t)d * 4;                 // dir 0
    const float* wb = cw + (size_t)(D_INNER + d) * 4;     // dir 1
    float accf = cb[d];
    float accb = cb[D_INNER + d];
#pragma unroll
    for (int k = 0; k < 4; ++k) {
        const int pf = l - 3 + k;
        const float xf = xz[(size_t)(pf >= 0 ? pf : 0) * (2 * D_INNER) + d];
        accf += (pf >= 0) ? wf[k] * xf : 0.f;
        const int pb = (L - 1 - l) + 3 - k;               // reversed-seq tap
        const float xb = xz[(size_t)(pb <= L - 1 ? pb : 0) * (2 * D_INNER) + d];
        accb += (pb <= L - 1) ? wb[k] * xb : 0.f;
    }
    xcf[idx] = accf * (1.f / (1.f + __expf(-accf)));      // silu
    xcb[idx] = accb * (1.f / (1.f + __expf(-accb)));
}

// ---------------------------------------------------------------------------
// Selective scan: thread owns one channel d, 16-entry state in registers,
// B_t/C_t broadcast via LDS each step.  y = (h·C + D*u) * silu(z).
// rev=1 -> z is read in reversed sequence order (backward direction).
// ---------------------------------------------------------------------------
__global__ void __launch_bounds__(256)
scan_kernel(const float* __restrict__ dt, const float* __restrict__ dbl,
            const float* __restrict__ xc, const float* __restrict__ xz,
            const float* __restrict__ A_log, const float* __restrict__ Dsk,
            float* __restrict__ y, int L, int rev)
{
    __shared__ float Bs[D_STATE];
    __shared__ float Cs[D_STATE];
    const int d = blockIdx.x * 256 + threadIdx.x;   // 6 blocks * 256 = 1536

    float Aa[D_STATE];
#pragma unroll
    for (int n = 0; n < D_STATE; ++n)
        Aa[n] = -__expf(A_log[(size_t)d * D_STATE + n]);
    const float Dv = Dsk[d];

    float h[D_STATE];
#pragma unroll
    for (int n = 0; n < D_STATE; ++n) h[n] = 0.f;

    for (int l = 0; l < L; ++l) {
        __syncthreads();
        if (threadIdx.x < D_STATE)
            Bs[threadIdx.x] = dbl[(size_t)l * 80 + DT_RANK + threadIdx.x];
        else if (threadIdx.x < 2 * D_STATE)
            Cs[threadIdx.x - D_STATE] = dbl[(size_t)l * 80 + DT_RANK + D_STATE + (threadIdx.x - D_STATE)];
        __syncthreads();

        const float dtl = dt[(size_t)l * D_INNER + d];
        const float u   = xc[(size_t)l * D_INNER + d];
        const float dtu = dtl * u;
        float accum = 0.f;
#pragma unroll
        for (int n = 0; n < D_STATE; ++n) {
            h[n] = h[n] * __expf(dtl * Aa[n]) + dtu * Bs[n];
            accum += h[n] * Cs[n];
        }
        const float yv = accum + Dv * u;
        const int zl = rev ? (L - 1 - l) : l;
        const float zv = xz[(size_t)zl * (2 * D_INNER) + D_INNER + d];
        y[(size_t)l * D_INNER + d] = yv * (zv / (1.f + __expf(-zv)));
    }
}

// ---------------------------------------------------------------------------
// Final rmsnorm at TOKPOS + 527-way classification head (single block).
// ---------------------------------------------------------------------------
__global__ void __launch_bounds__(256)
final_head(const float* __restrict__ hidden, const float* __restrict__ residual,
           const float* __restrict__ nfw, const float* __restrict__ hw,
           const float* __restrict__ hb, float* __restrict__ out)
{
    __shared__ float feat[D_MODEL];
    __shared__ float red[256];
    const int l = TOKPOS;
    float ss = 0.f;
    for (int c = threadIdx.x; c < D_MODEL; c += 256) {
        const float r = residual[(size_t)l * D_MODEL + c] + hidden[(size_t)l * D_MODEL + c];
        feat[c] = r;
        ss += r * r;
    }
    red[threadIdx.x] = ss;
    __syncthreads();
    for (int s = 128; s > 0; s >>= 1) {
        if (threadIdx.x < s) red[threadIdx.x] += red[threadIdx.x + s];
        __syncthreads();
    }
    const float scale = rsqrtf(red[0] / (float)D_MODEL + 1e-5f);
    for (int c = threadIdx.x; c < D_MODEL; c += 256)
        feat[c] = feat[c] * scale * nfw[c];
    __syncthreads();
    for (int cls = threadIdx.x; cls < NUM_CLASSES; cls += 256) {
        float a = hb[cls];
        const float* wrow = hw + (size_t)cls * D_MODEL;
        for (int c = 0; c < D_MODEL; ++c) a += feat[c] * wrow[c];
        out[cls] = a;
    }
}

// ---------------------------------------------------------------------------
extern "C" void kernel_launch(void* const* d_in, const int* in_sizes, int n_in,
                              void* d_out, int out_size, void* d_ws, size_t ws_size,
                              hipStream_t stream)
{
    (void)in_sizes; (void)n_in; (void)out_size; (void)ws_size;
    const float* x         = (const float*)d_in[0];
    const float* patch_w   = (const float*)d_in[1];
    const float* patch_b   = (const float*)d_in[2];
    const float* cls_token = (const float*)d_in[3];
    const float* pos_embed = (const float*)d_in[4];
    const float* in_proj_w = (const float*)d_in[5];
    const float* conv_w    = (const float*)d_in[6];
    const float* conv_b    = (const float*)d_in[7];
    const float* xproj_w   = (const float*)d_in[8];
    const float* dtproj_w  = (const float*)d_in[9];
    const float* dtproj_b  = (const float*)d_in[10];
    const float* A_log     = (const float*)d_in[11];
    const float* D_skip    = (const float*)d_in[12];
    const float* out_projw = (const float*)d_in[13];
    const float* norm_w    = (const float*)d_in[14];
    const float* norm_f_w  = (const float*)d_in[15];
    const float* head_w    = (const float*)d_in[16];
    const float* head_b    = (const float*)d_in[17];

    float* ws = (float*)d_ws;
    size_t off = 0;
    auto alloc = [&](size_t n) { float* p = ws + off; off += n; return p; };
    float* Ap       = alloc((size_t)SEQ * 256);
    float* tokens   = alloc((size_t)SEQ * D_MODEL);
    float* hidden   = alloc((size_t)LTOT * D_MODEL);
    float* residual = alloc((size_t)LTOT * D_MODEL);
    float* hbuf     = alloc((size_t)LTOT * D_MODEL);
    float* xzb      = alloc((size_t)LTOT * 2 * D_INNER);
    float* xcf      = alloc((size_t)LTOT * D_INNER);
    float* xcb      = alloc((size_t)LTOT * D_INNER);
    float* dblf     = alloc((size_t)LTOT * 80);
    float* dblb     = alloc((size_t)LTOT * 80);
    float* dtf      = alloc((size_t)LTOT * D_INNER);
    float* dtb      = alloc((size_t)LTOT * D_INNER);
    float* yf       = alloc((size_t)LTOT * D_INNER);
    float* yb       = alloc((size_t)LTOT * D_INNER);

    auto gemm = [&](const float* A, const float* W, const float* bias, float* C,
                    int M, int N, int K, int lda, int modeE) {
        const int strips = ((M + 15) / 16) * ((N + 63) / 64);
        const int blocks = (strips + 3) / 4;
        gemm_bf16_wmma<0><<<blocks, 128, 0, stream>>>(A, nullptr, W, bias, C,
                                                      M, N, K, lda, modeE);
    };
    auto gemm_comb = [&](const float* A, const float* Arev, const float* W,
                         float* C, int M, int N, int K, int lda) {
        const int strips = ((M + 15) / 16) * ((N + 63) / 64);
        const int blocks = (strips + 3) / 4;
        gemm_bf16_wmma<1><<<blocks, 128, 0, stream>>>(A, Arev, W, nullptr, C,
                                                      M, N, K, lda, 0);
    };

    // Patch embed
    im2col_patch<<<(SEQ * 256) / 256, 256, 0, stream>>>(x, Ap);
    gemm(Ap, patch_w, patch_b, tokens, SEQ, D_MODEL, 256, 256, 1);
    assemble_tokens<<<(LTOT * D_MODEL + 255) / 256, 256, 0, stream>>>(
        tokens, cls_token, pos_embed, hidden, residual);

    const int L = LTOT;
    for (int layer = 0; layer < DEPTH; ++layer) {
        resid_rmsnorm<<<L, 256, 0, stream>>>(hidden, residual,
                                             norm_w + (size_t)layer * D_MODEL, hbuf);
        gemm(hbuf, in_proj_w + (size_t)layer * 2 * D_INNER * D_MODEL,
             nullptr, xzb, L, 2 * D_INNER, D_MODEL, D_MODEL, 0);
        conv_silu<<<(L * D_INNER + 255) / 256, 256, 0, stream>>>(
            xzb, conv_w + (size_t)layer * 2 * D_INNER * 4,
            conv_b + (size_t)layer * 2 * D_INNER, xcf, xcb, L);

        const float* xwf = xproj_w + (size_t)(layer * 2 + 0) * 80 * D_INNER;
        const float* xwb = xproj_w + (size_t)(layer * 2 + 1) * 80 * D_INNER;
        gemm(xcf, xwf, nullptr, dblf, L, 80, D_INNER, D_INNER, 0);
        gemm(xcb, xwb, nullptr, dblb, L, 80, D_INNER, D_INNER, 0);

        const float* dwf = dtproj_w + (size_t)(layer * 2 + 0) * D_INNER * DT_RANK;
        const float* dwb = dtproj_w + (size_t)(layer * 2 + 1) * D_INNER * DT_RANK;
        const float* dbf = dtproj_b + (size_t)(layer * 2 + 0) * D_INNER;
        const float* dbb = dtproj_b + (size_t)(layer * 2 + 1) * D_INNER;
        gemm(dblf, dwf, dbf, dtf, L, D_INNER, DT_RANK, 80, 2);
        gemm(dblb, dwb, dbb, dtb, L, D_INNER, DT_RANK, 80, 2);

        scan_kernel<<<D_INNER / 256, 256, 0, stream>>>(
            dtf, dblf, xcf, xzb,
            A_log + (size_t)(layer * 2 + 0) * D_INNER * D_STATE,
            D_skip + (size_t)(layer * 2 + 0) * D_INNER, yf, L, 0);
        scan_kernel<<<D_INNER / 256, 256, 0, stream>>>(
            dtb, dblb, xcb, xzb,
            A_log + (size_t)(layer * 2 + 1) * D_INNER * D_STATE,
            D_skip + (size_t)(layer * 2 + 1) * D_INNER, yb, L, 1);

        gemm_comb(yf, yb, out_projw + (size_t)layer * D_MODEL * D_INNER,
                  hidden, L, D_MODEL, D_INNER, D_INNER);
    }

    final_head<<<1, 256, 0, stream>>>(hidden, residual, norm_f_w, head_w, head_b,
                                      (float*)d_out);
}
